// Custom_Linear_75393855914392
// MI455X (gfx1250) — 8-bit matrix operand order repaired and checked by reading, compiled, not run on hardware
//
#include <hip/hip_runtime.h>
#include <cstdint>
#include <cstddef>


typedef int v8i __attribute__((ext_vector_type(8)));
typedef int v4i __attribute__((ext_vector_type(4)));
typedef int v2i __attribute__((ext_vector_type(2)));

constexpr int S = 8192;
constexpr int K = 1024;
constexpr int O = 4096;

constexpr int BM = 128;
constexpr int BN = 256;
constexpr int BK = 64;

__global__ __launch_bounds__(256) void requant_prep_kernel(
    const float* __restrict__ sx, const float* __restrict__ sw,
    const float* __restrict__ sy, float* __restrict__ requant,
    float* __restrict__ out_tail) {
  const int s = blockIdx.x * blockDim.x + threadIdx.x;
  if (s < S) {
    const float ys = sy[s];
    requant[s] = sx[0] * sw[0] / ys;
    out_tail[s] = ys;
  }
}

__global__ __launch_bounds__(256) void int8_gemm_wmma_kernel(
    const int8_t* __restrict__ X,
    const int8_t* __restrict__ W,
    const float* __restrict__ requant,
    float* __restrict__ out) {
  const int lane = threadIdx.x & 31;
  const int wave = threadIdx.x >> 5;
  const int waveM = wave >> 2;
  const int waveN = wave & 3;
  const int half  = lane >> 4;
  const int l16   = lane & 15;

  const int rowBase = blockIdx.y * BM + waveM * 64;
  const int colBase = blockIdx.x * BN + waveN * 64;

  const int8_t* xrow[4];
#pragma unroll
  for (int mi = 0; mi < 4; ++mi)
    xrow[mi] = X + (size_t)(rowBase + mi * 16 + l16) * K;

  const int8_t* wrow[4];
#pragma unroll
  for (int ni = 0; ni < 4; ++ni)
    wrow[ni] = W + (size_t)(colBase + ni * 16 + l16) * K;

  const int aK = half * 16;
  const int bK = half * 16;

  v8i acc[4][4];
#pragma unroll
  for (int mi = 0; mi < 4; ++mi)
#pragma unroll
    for (int ni = 0; ni < 4; ++ni)
      acc[mi][ni] = v8i{};

  for (int k = 0; k < K; k += BK) {
    v8i a[4];
#pragma unroll
    for (int mi = 0; mi < 4; ++mi) {
      const int8_t* p = xrow[mi] + k + aK;
      const v2i t0 = *(const v2i*)(p);
      const v2i t1 = *(const v2i*)(p + 8);
      const v2i t2 = *(const v2i*)(p + 32);
      const v2i t3 = *(const v2i*)(p + 40);
      v8i av;
      av[0] = t0.x; av[1] = t0.y;
      av[2] = t1.x; av[3] = t1.y;
      av[4] = t2.x; av[5] = t2.y;
      av[6] = t3.x; av[7] = t3.y;
      a[mi] = av;
    }

    v8i b[4];
#pragma unroll
    for (int ni = 0; ni < 4; ++ni) {
      const int8_t* p = wrow[ni] + k + bK;
      const v4i t0 = *(const v4i*)(p);
      const v4i t1 = *(const v4i*)(p + 32);
      v8i bv;
      bv[0] = t0.x; bv[1] = t0.y; bv[2] = t0.z; bv[3] = t0.w;
      bv[4] = t1.x; bv[5] = t1.y; bv[6] = t1.z; bv[7] = t1.w;
      b[ni] = bv;
    }

    if (k + BK < K) {
#pragma unroll
      for (int mi = 0; mi < 4; ++mi)
        __builtin_prefetch((const void*)(xrow[mi] + k + BK + aK), 0, 3);
#pragma unroll
      for (int ni = 0; ni < 4; ++ni)
        __builtin_prefetch((const void*)(wrow[ni] + k + BK + bK), 0, 3);
    }

#pragma unroll
    for (int mi = 0; mi < 4; ++mi)
#pragma unroll
      for (int ni = 0; ni < 4; ++ni)
        acc[mi][ni] = __builtin_amdgcn_wmma_i32_16x16x64_iu8(
             true, a[mi],  true, b[ni], acc[mi][ni],
             false,  false);
  }

#pragma unroll
  for (int mi = 0; mi < 4; ++mi) {
    float rs[8];
    int   srow[8];
#pragma unroll
    for (int r = 0; r < 8; ++r) {
      srow[r] = rowBase + mi * 16 + r + half * 8;
      rs[r] = requant[srow[r]];
    }
#pragma unroll
    for (int ni = 0; ni < 4; ++ni) {
      const int ocol = colBase + ni * 16 + l16;
#pragma unroll
      for (int r = 0; r < 8; ++r) {
        float v = (float)acc[mi][ni][r] * rs[r];
        v = rintf(v);
        v = fminf(fmaxf(v, -128.0f), 127.0f);
        __builtin_nontemporal_store(v, &out[(size_t)srow[r] * O + ocol]);
      }
    }
  }
}

extern "C" void kernel_launch(void* const* d_in, const int* in_sizes, int n_in,
                              void* d_out, int out_size, void* d_ws,
                              size_t ws_size, hipStream_t stream) {
  const int8_t* x  = (const int8_t*)d_in[0];
  const int8_t* w  = (const int8_t*)d_in[1];
  const float*  sx = (const float*)d_in[2];
  const float*  sw = (const float*)d_in[3];
  const float*  sy = (const float*)d_in[4];

  float* out      = (float*)d_out;
  float* out_tail = out + (size_t)S * O;
  float* requant  = (float*)d_ws;

  requant_prep_kernel<<<(S + 255) / 256, 256, 0, stream>>>(sx, sw, sy, requant,
                                                           out_tail);

  dim3 grid(O / BN, S / BM);
  int8_gemm_wmma_kernel<<<grid, 256, 0, stream>>>(x, w, requant, out);
}
